// GATv2_29351806501504
// MI455X (gfx1250) — matrix operand order repaired and checked by reading, not compiled, not run on hardware
//
#include <hip/hip_runtime.h>
#include <math.h>

// ---------------------------------------------------------------------------
// GATv2 3-layer forward for MI455X (gfx1250, wave32).
//   N=50000 nodes, E=800000 edges (+N self loops), F=128.
//   GEMMs via v_wmma_f32_16x16x32_bf16 (fp32 accum), 16 WMMAs per wave with
//   A-fragment reuse across a 16x64 output strip. Edge phase via
//   wave-per-(edge,head) reductions + L2-resident gathers/atomics
//   (xl/xr/agg ~77MB/layer << 192MB L2).
// ---------------------------------------------------------------------------

#define NNODES    50000
#define KDIM      128        // inner dim of every GEMM in this model
#define NEG_SLOPE 0.2f

typedef __attribute__((ext_vector_type(16))) __bf16 v16bf;
typedef __attribute__((ext_vector_type(8)))  float  v8f;

__device__ __forceinline__ __bf16 f2bf(float f) { return static_cast<__bf16>(f); }

// Order-preserving monotone map float -> uint32 (for atomicMax on signed floats)
__device__ __forceinline__ unsigned f2ord(float f) {
    unsigned u = __float_as_uint(f);
    return (u & 0x80000000u) ? ~u : (u | 0x80000000u);
}
__device__ __forceinline__ float ord2f(unsigned u) {
    return (u & 0x80000000u) ? __uint_as_float(u & 0x7fffffffu)
                             : __uint_as_float(~u);
}

// ---------------------------------------------------------------------------
// Dual GEMM: XL = X @ Wl, XR = X @ Wr.  X:[rows,128], W:[128,Fout] row-major.
// One wave per 16x64 output strip (4 col tiles): A fragment reused 4x,
// K fully unrolled -> 16 v_wmma per wave. Grid tiles output exactly
// (rows % 16 == 0, Fout % 64 == 0), so no store guards and EXEC stays all-1s.
//
// CDNA5 16-bit A (16x32) layout: lanes 0-15 hold M=lane, elems 0..7 = K 0..7,
// elems 8..15 = K 16..23; lanes 16-31 hold K 8..15 / 24..31.
// B (32x16) as loaded here: lane L holds column N=L%16; lanes 0-15 cover K 0..15,
// lanes 16-31 cover K 16..31 (K ascending, 2 bf16 per VGPR); A is loaded in this same K order.
// C/D (16x16 f32, 8 VGPRs): VGPR v -> row v + 8*(lane>=16), col = lane%16.
// ---------------------------------------------------------------------------
__global__ void gatv2_dual_gemm_wmma(const float* __restrict__ X,
                                     const float* __restrict__ Wl,
                                     const float* __restrict__ Wr,
                                     float* __restrict__ XL,
                                     float* __restrict__ XR,
                                     int Fout)
{
    const int lane = threadIdx.x & 31;
    const int half = lane >> 4;   // 0: lanes 0-15, 1: lanes 16-31
    const int lr   = lane & 15;

    const int ncg       = Fout >> 6;                 // 64-col groups per matrix
    const bool is_left  = ((int)blockIdx.y < ncg);
    const int  grp      = is_left ? blockIdx.y : (blockIdx.y - ncg);
    const float* W      = is_left ? Wl : Wr;
    float*       Out    = is_left ? XL : XR;
    const int colbase   = grp * 64;

    const int row = blockIdx.x * 16 + lr;            // rows % 16 == 0, no clamp
    const float* Arow = X + (size_t)row * KDIM;

    v8f acc0 = {}, acc1 = {}, acc2 = {}, acc3 = {};

#pragma unroll
    for (int k0 = 0; k0 < KDIM; k0 += 32) {
        // ---- A fragment (shared by all 4 column tiles) ----
        v16bf a;
        const int ab = k0 + 16 * half;
#pragma unroll
        for (int i = 0; i < 8; ++i) {
            a[i]     = f2bf(Arow[ab + i]);
            a[i + 8] = f2bf(Arow[ab + 8 + i]);
        }
        const int kb = k0 + 16 * half;
        const float* Wk = W + (size_t)kb * Fout + colbase + lr;

        // ---- 4 B fragments / 4 WMMAs, reusing A ----
        v16bf b;
#pragma unroll
        for (int i = 0; i < 16; ++i) b[i] = f2bf(Wk[(size_t)i * Fout]);
        acc0 = __builtin_amdgcn_wmma_f32_16x16x32_bf16(false, a, false, b,
                                                       (short)0, acc0, false, false);
#pragma unroll
        for (int i = 0; i < 16; ++i) b[i] = f2bf(Wk[(size_t)i * Fout + 16]);
        acc1 = __builtin_amdgcn_wmma_f32_16x16x32_bf16(false, a, false, b,
                                                       (short)0, acc1, false, false);
#pragma unroll
        for (int i = 0; i < 16; ++i) b[i] = f2bf(Wk[(size_t)i * Fout + 32]);
        acc2 = __builtin_amdgcn_wmma_f32_16x16x32_bf16(false, a, false, b,
                                                       (short)0, acc2, false, false);
#pragma unroll
        for (int i = 0; i < 16; ++i) b[i] = f2bf(Wk[(size_t)i * Fout + 48]);
        acc3 = __builtin_amdgcn_wmma_f32_16x16x32_bf16(false, a, false, b,
                                                       (short)0, acc3, false, false);
    }

    // ---- store 16x64 strip (rows v + 8*half, cols colbase + t*16 + lr) ----
    float* Ob = Out + (size_t)(blockIdx.x * 16 + 8 * half) * Fout + colbase + lr;
#pragma unroll
    for (int v = 0; v < 8; ++v) {
        float* Or = Ob + (size_t)v * Fout;
        Or[0]  = acc0[v];
        Or[16] = acc1[v];
        Or[32] = acc2[v];
        Or[48] = acc3[v];
    }
}

// ---------------------------------------------------------------------------
// Edge scores: one wave per (edge, head).
//   e[eh] = att[h] . leaky_relu(xl[src] + xr[dst]) ; segment-max into nmax.
// ---------------------------------------------------------------------------
__global__ void gatv2_edge_scores(const float* __restrict__ xl,
                                  const float* __restrict__ xr,
                                  const int*   __restrict__ ei,
                                  const float* __restrict__ att,
                                  float* __restrict__ esc,
                                  unsigned* __restrict__ nmax,
                                  int E, int ET, int H, int D)
{
    long long gid = (long long)blockIdx.x * blockDim.x + threadIdx.x;
    long long wid = gid >> 5;                       // uniform within wave
    int lane = (int)(gid & 31);
    if (wid >= (long long)ET * H) return;

    int e = (int)(wid / H);
    int h = (int)(wid % H);
    int s  = (e < E) ? ei[e]     : (e - E);         // self-loop tail
    int dt = (e < E) ? ei[E + e] : (e - E);

    const int F = H * D;
    const float* xls = xl + (size_t)s  * F + h * D;
    const float* xrd = xr + (size_t)dt * F + h * D;
    const float* at  = att + h * D;

    float p = 0.f;
    for (int d = lane; d < D; d += 32) {
        float m = xls[d] + xrd[d];
        m = (m > 0.f) ? m : NEG_SLOPE * m;
        p += m * at[d];
    }
#pragma unroll
    for (int off = 16; off > 0; off >>= 1)
        p += __shfl_down(p, off, 32);
    if (lane == 0) {
        esc[(size_t)e * H + h] = p;
        atomicMax(&nmax[(size_t)dt * H + h], f2ord(p));
    }
}

// ---------------------------------------------------------------------------
// Edge weights (in place): w = exp(e - nmax[dst]); denom[dst] += w.
// ---------------------------------------------------------------------------
__global__ void gatv2_edge_weights(float* __restrict__ esc,
                                   const unsigned* __restrict__ nmax,
                                   float* __restrict__ denom,
                                   const int* __restrict__ ei,
                                   int E, int ET, int H)
{
    long long gid = (long long)blockIdx.x * blockDim.x + threadIdx.x;
    if (gid >= (long long)ET * H) return;
    int e = (int)(gid / H);
    int h = (int)(gid % H);
    int dt = (e < E) ? ei[E + e] : (e - E);
    float w = expf(esc[gid] - ord2f(nmax[(size_t)dt * H + h]));
    esc[gid] = w;
    atomicAdd(&denom[(size_t)dt * H + h], w);
}

// ---------------------------------------------------------------------------
// Aggregate: one wave per edge; agg[dst] += (w/denom[dst]) * xl[src].
// ---------------------------------------------------------------------------
__global__ void gatv2_aggregate(const float* __restrict__ xl,
                                const float* __restrict__ esc,
                                const float* __restrict__ denom,
                                const int*   __restrict__ ei,
                                float* __restrict__ agg,
                                int E, int ET, int H, int D)
{
    long long gid = (long long)blockIdx.x * blockDim.x + threadIdx.x;
    long long wid = gid >> 5;
    int lane = (int)(gid & 31);
    if (wid >= ET) return;

    int e  = (int)wid;
    int s  = (e < E) ? ei[e]     : (e - E);
    int dt = (e < E) ? ei[E + e] : (e - E);
    const int F = H * D;
    for (int f = lane; f < F; f += 32) {
        int h = f / D;
        float alpha = esc[(size_t)e * H + h] /
                      (denom[(size_t)dt * H + h] + 1e-16f);
        atomicAdd(&agg[(size_t)dt * F + f], alpha * xl[(size_t)s * F + f]);
    }
}

// ---------------------------------------------------------------------------
// Bias (+ optional ELU):  out = elu?(in + b)
// ---------------------------------------------------------------------------
__global__ void gatv2_bias_act(const float* __restrict__ in,
                               float* __restrict__ out,
                               const float* __restrict__ b,
                               int n, int F, int do_elu)
{
    long long gid = (long long)blockIdx.x * blockDim.x + threadIdx.x;
    if (gid >= (long long)n * F) return;
    int f = (int)(gid % F);
    float v = in[gid] + b[f];
    if (do_elu) v = (v > 0.f) ? v : expm1f(v);
    out[gid] = v;
}

// ---------------------------------------------------------------------------
// Host launcher
// ---------------------------------------------------------------------------
extern "C" void kernel_launch(void* const* d_in, const int* in_sizes, int n_in,
                              void* d_out, int out_size, void* d_ws, size_t ws_size,
                              hipStream_t stream)
{
    const float* x   = (const float*)d_in[0];
    const int*   ei  = (const int*)  d_in[1];
    const float* Wl0 = (const float*)d_in[2];
    const float* Wr0 = (const float*)d_in[3];
    const float* at0 = (const float*)d_in[4];
    const float* b0  = (const float*)d_in[5];
    const float* Wl1 = (const float*)d_in[6];
    const float* Wr1 = (const float*)d_in[7];
    const float* at1 = (const float*)d_in[8];
    const float* b1  = (const float*)d_in[9];
    const float* Wl2 = (const float*)d_in[10];
    const float* Wr2 = (const float*)d_in[11];
    const float* at2 = (const float*)d_in[12];
    const float* b2  = (const float*)d_in[13];

    const int N  = NNODES;
    const int E  = in_sizes[1] / 2;      // 800000
    const int ET = E + N;                // + self loops

    // Workspace layout (floats):
    //   xl [N*128] | xr [N*128] | esc [ET*4] | nmax [N*4](u32) | denom [N*4] | agg [N*128]
    float*    xl    = (float*)d_ws;
    float*    xr    = xl  + (size_t)N * 128;
    float*    esc   = xr  + (size_t)N * 128;
    unsigned* nmax  = (unsigned*)(esc + (size_t)ET * 4);
    float*    denom = (float*)(nmax + (size_t)N * 4);
    float*    agg   = denom + (size_t)N * 4;   // also next layer's input

    auto run_layer = [&](const float* X, const float* Wl, const float* Wr,
                         const float* att, const float* bias,
                         int H, int D, float* layer_out, int do_elu) {
        const int Fout = H * D;
        // 1) dual GEMM (WMMA): xl = X@Wl, xr = X@Wr
        //    grid.y: [Wl groups | Wr groups], 64 cols per wave
        dim3 ggrid(N / 16, 2 * (Fout / 64));
        gatv2_dual_gemm_wmma<<<ggrid, 32, 0, stream>>>(X, Wl, Wr, xl, xr, Fout);
        // 2) zero segment state + accumulator (X no longer needed after GEMM)
        hipMemsetAsync(nmax,  0, (size_t)N * H * sizeof(unsigned), stream);
        hipMemsetAsync(denom, 0, (size_t)N * H * sizeof(float),    stream);
        hipMemsetAsync(agg,   0, (size_t)N * Fout * sizeof(float), stream);
        // 3) edge scores + segment max
        {
            long long thr = (long long)ET * H * 32;
            gatv2_edge_scores<<<(unsigned)((thr + 255) / 256), 256, 0, stream>>>(
                xl, xr, ei, att, esc, nmax, E, ET, H, D);
        }
        // 4) softmax numerator + denominator
        {
            long long thr = (long long)ET * H;
            gatv2_edge_weights<<<(unsigned)((thr + 255) / 256), 256, 0, stream>>>(
                esc, nmax, denom, ei, E, ET, H);
        }
        // 5) weighted aggregation
        {
            long long thr = (long long)ET * 32;
            gatv2_aggregate<<<(unsigned)((thr + 255) / 256), 256, 0, stream>>>(
                xl, esc, denom, ei, agg, E, ET, H, D);
        }
        // 6) bias + activation
        {
            long long thr = (long long)N * Fout;
            gatv2_bias_act<<<(unsigned)((thr + 255) / 256), 256, 0, stream>>>(
                agg, layer_out, bias, N, Fout, do_elu);
        }
    };

    // Layer 0: 128 -> 4x32, concat, ELU            (output in agg, in place)
    run_layer(x,   Wl0, Wr0, at0, b0, 4, 32, agg, 1);
    // Layer 1: 128 -> 4x32, concat, ELU            (reads agg, writes agg)
    run_layer(agg, Wl1, Wr1, at1, b1, 4, 32, agg, 1);
    // Layer 2: 128 -> 1x64, H=1 so mean==identity; bias only, write d_out
    run_layer(agg, Wl2, Wr2, at2, b2, 1, 64, (float*)d_out, 0);
}
